// DSAttention_53601191854569
// MI455X (gfx1250) — matrix operand order repaired and checked by reading, not compiled, not run on hardware
//
#include <hip/hip_runtime.h>

// DSAttention (causal, tau/delta-modulated) fused flash-attention for gfx1250.
// B=8, L=1024, H=16, D=64, fp32 in/out; bf16 WMMA (v_wmma_f32_16x16x32_bf16)
// with f32 accumulation. One workgroup (128 thr / 4 waves) = 64 queries of one
// (b,h); each wave owns a 16-row query tile; inner loop over 32-key blocks.
//
// v4: softmax in log2 domain (bare v_exp_f32, no per-exp mul); reduction
// combines via inline v_max_num_f32 (kills the bitcast-induced canonicalize).

#define B_ 8
#define L_ 1024
#define H_ 16
#define D_ 64

typedef __attribute__((ext_vector_type(16))) __bf16 v16bf;
typedef __attribute__((ext_vector_type(8)))  __bf16 v8bf;
typedef __attribute__((ext_vector_type(8)))  float  v8f;
typedef __attribute__((ext_vector_type(4)))  float  v4f;

// bf16 strides (elements). 72 bf16 = 144B row step = 36-bank step,
// gcd(36,64)=4 -> 16 lanes hit 16 distinct start banks; 16B aligned chunks.
#define SKH 72   // K tile: 32 rows (keys) x 64 dims
#define SVH 72   // V^T tile: 64 rows (dims) x 32 keys
#define SPH 40   // P tile per wave: 16 rows x 32 cols (80B step, 20-bank)

// ds_swizzle_b32 group-of-32 xor patterns: offset = xor<<10 | and_mask(0x1f)
#define SWZ_X1 0x041F
#define SWZ_X2 0x081F
#define SWZ_X4 0x101F
#define SWZ_X8 0x201F
#define SWZF(x, imm) \
    __int_as_float(__builtin_amdgcn_ds_swizzle(__float_as_int(x), (imm)))

#if __has_builtin(__builtin_amdgcn_exp2f)
#define EXP2F(x) __builtin_amdgcn_exp2f(x)
#else
#define EXP2F(x) exp2f(x)
#endif

// single v_max_num_f32; avoids LLVM's canonicalize on bitcast swizzle results
__device__ __forceinline__ float vmax2(float a, float b) {
    float d;
    asm("v_max_num_f32 %0, %1, %2" : "=v"(d) : "v"(a), "v"(b));
    return d;
}

__device__ __forceinline__ float half_reduce_max(float x) {
    x = vmax2(x, SWZF(x, SWZ_X1));
    x = vmax2(x, SWZF(x, SWZ_X2));
    x = vmax2(x, SWZF(x, SWZ_X4));
    x = vmax2(x, SWZF(x, SWZ_X8));
    return x;   // max over each 16-lane half (xor masks < 16 stay in-half)
}
__device__ __forceinline__ float half_reduce_sum(float x) {
    x += SWZF(x, SWZ_X1);
    x += SWZF(x, SWZ_X2);
    x += SWZF(x, SWZ_X4);
    x += SWZF(x, SWZ_X8);
    return x;
}

__device__ __forceinline__ v16bf ld_bf16x16(const __bf16* p) {
    v8bf lo = *(const v8bf*)p;
    v8bf hi = *(const v8bf*)(p + 16);
    return __builtin_shufflevector(lo, hi, 0,1,2,3,4,5,6,7,8,9,10,11,12,13,14,15);
}

__launch_bounds__(128)
__global__ void dsattn_fwd_kernel(const float* __restrict__ Q,
                                  const float* __restrict__ K,
                                  const float* __restrict__ V,
                                  const float* __restrict__ tau,
                                  const float* __restrict__ delta,
                                  float* __restrict__ O)
{
    __shared__ __bf16 sKh [32 * SKH];
    __shared__ __bf16 sVth[64 * SVH];
    __shared__ __bf16 sPh [4 * 16 * SPH];

    const int qblk = blockIdx.x;          // 16 query blocks of 64
    const int h    = blockIdx.y;
    const int b    = blockIdx.z;
    const int tid  = threadIdx.x;
    const int wave = tid >> 5;
    const int lid  = tid & 31;
    const int half = lid >> 4;            // lane half (wave32 WMMA layout)
    const int hn   = lid & 15;

    const int q0 = qblk * 64 + wave * 16;        // this wave's 16 query rows
    // log2-domain folded scale: logit2 = (scale*log2e*tau)*qk + (scale*log2e)*delta
    const float c2  = 0.125f * 1.44269504088896340736f;  // scale * log2(e)
    const float st2 = c2 * tau[b];

    // ---- Q tile (16x64) as two A-fragments (16x32 bf16 each) -------------
    // A-layout (16-bit, 16x32): lane M = lane&15; element i ->
    // K = (i>>3)*16 + half*8 + (i&7)  => two contiguous 8-float runs.
    v16bf qa[2];
    {
        const float* qrow = Q + (((size_t)b * L_ + (q0 + hn)) * H_ + h) * D_;
        #pragma unroll
        for (int f = 0; f < 2; ++f) {
            const float* p = qrow + f * 32 + half * 8;
            v4f a0 = *(const v4f*)(p);
            v4f a1 = *(const v4f*)(p + 4);
            v4f a2 = *(const v4f*)(p + 16);
            v4f a3 = *(const v4f*)(p + 20);
            v16bf q;
            #pragma unroll
            for (int e = 0; e < 4; ++e) {
                q[e]      = (__bf16)a0[e];
                q[4 + e]  = (__bf16)a1[e];
                q[8 + e]  = (__bf16)a2[e];
                q[12 + e] = (__bf16)a3[e];
            }
            qa[f] = q;
        }
    }

    v8f acc[4] = {v8f{}, v8f{}, v8f{}, v8f{}};   // 16 queries x 64 dims, f32
    float mrow[8], lrow[8];                      // log2-domain m; per-lane l
    #pragma unroll
    for (int r = 0; r < 8; ++r) { mrow[r] = -1e30f; lrow[r] = 0.0f; }

    __bf16* sPw = sPh + wave * 16 * SPH;
    const int nkb = 2 * qblk + 2;                // causal: keys <= q0_block+63

    // ---- per-key-block body (MASKED folded after inlining) ----------------
    auto block_body = [&](int key_base, bool MASKED) {
        // scores: two 16x16 tiles, each = 2 WMMAs over dims.
        // B-layout (16-bit, 32x16): lane N = lane&15; element i -> K = (i>>3)*16 + half*8 + (i&7)
        // => two contiguous 8-bf16 runs per fragment.
        v8f sc[2];
        #pragma unroll
        for (int t = 0; t < 2; ++t) {
            const __bf16* kr = sKh + (16 * t + hn) * SKH + half * 8;
            v16bf kb0 = ld_bf16x16(kr);
            v16bf kb1 = ld_bf16x16(kr + 32);
            v8f s = {};
            s = __builtin_amdgcn_wmma_f32_16x16x32_bf16(false, qa[0], false, kb0,
                                                        (short)0, s, false, false);
            s = __builtin_amdgcn_wmma_f32_16x16x32_bf16(false, qa[1], false, kb1,
                                                        (short)0, s, false, false);
            sc[t] = s;
        }

        // logits (log2 domain) + online softmax.
        // C-layout: reg r, lane -> M = r + 8*half (query), N = lane&15 (key).
        const float sdl0 = c2 * delta[(size_t)b * L_ + key_base + hn];
        const float sdl1 = c2 * delta[(size_t)b * L_ + key_base + 16 + hn];
        const int   kn0  = key_base + hn;
        const int   kn1  = key_base + 16 + hn;
        #pragma unroll
        for (int r = 0; r < 8; ++r) {
            float lg0 = fmaf(st2, sc[0][r], sdl0);
            float lg1 = fmaf(st2, sc[1][r], sdl1);
            if (MASKED) {
                const int qm = q0 + r + 8 * half;
                if (kn0 > qm) lg0 = -1e30f;
                if (kn1 > qm) lg1 = -1e30f;
            }
            const float mloc = half_reduce_max(fmaxf(lg0, lg1));
            const float mnew = vmax2(mrow[r], mloc);
            const float al   = EXP2F(mrow[r] - mnew);  // bare v_exp_f32
            const float p0   = EXP2F(lg0 - mnew);
            const float p1   = EXP2F(lg1 - mnew);
            lrow[r] = lrow[r] * al + (p0 + p1);   // per-lane; reduce at end
            mrow[r] = mnew;
            acc[0][r] *= al; acc[1][r] *= al; acc[2][r] *= al; acc[3][r] *= al;

            // stage P as bf16 (C-layout -> row-major); wave-private tile,
            // DS ops are in-order within a wave => no barrier needed.
            sPw[(r + 8 * half) * SPH + hn]      = (__bf16)p0;
            sPw[(r + 8 * half) * SPH + 16 + hn] = (__bf16)p1;
        }

        // P as A-fragment: rows = lane&15, two contiguous 8-bf16 runs.
        const __bf16* pr = sPw + hn * SPH + half * 8;
        v8bf plo = *(const v8bf*)(pr);
        v8bf phi = *(const v8bf*)(pr + 16);
        v16bf pa = __builtin_shufflevector(plo, phi,
                       0,1,2,3,4,5,6,7,8,9,10,11,12,13,14,15);

        // PV: 4 WMMAs (D=64 -> 4 N-tiles); V^T staged so B-fragments are
        // two contiguous 8-bf16 runs.
        #pragma unroll
        for (int n4 = 0; n4 < 4; ++n4) {
            v16bf vb = ld_bf16x16(sVth + (16 * n4 + hn) * SVH + half * 8);
            acc[n4] = __builtin_amdgcn_wmma_f32_16x16x32_bf16(false, pa, false, vb,
                                                              (short)0, acc[n4],
                                                              false, false);
        }
    };

    for (int j = 0; j < nkb; ++j) {
        const int key_base = j * 32;

        __syncthreads();
        // ---- cooperative stage: K bf16 row-major, V bf16 transposed ------
        {
            const int row = tid >> 2;            // 0..31 (key within block)
            const int col = (tid & 3) * 16;      // 0/16/32/48 (dim)
            const size_t gbase = (((size_t)b * L_ + key_base + row) * H_ + h) * D_ + col;
            const float* kg = K + gbase;
            const float* vg = V + gbase;
            v4f kt[4], vt[4];
            #pragma unroll
            for (int e = 0; e < 4; ++e) {
                kt[e] = *(const v4f*)(kg + 4 * e);
                vt[e] = *(const v4f*)(vg + 4 * e);
            }
            // K: 16 contiguous bf16 per thread (two 16B chunks)
            v16bf kv;
            #pragma unroll
            for (int e = 0; e < 16; ++e) kv[e] = (__bf16)kt[e >> 2][e & 3];
            __bf16* kd = sKh + row * SKH + col;
            *(v8bf*)(kd)     = __builtin_shufflevector(kv, kv, 0,1,2,3,4,5,6,7);
            *(v8bf*)(kd + 8) = __builtin_shufflevector(kv, kv, 8,9,10,11,12,13,14,15);
            // V transposed: scatter bf16
            #pragma unroll
            for (int e = 0; e < 16; ++e)
                sVth[(col + e) * SVH + row] = (__bf16)vt[e >> 2][e & 3];

            // prefetch next key block into cache (global_prefetch_b8)
            if (j + 1 < nkb) {
                __builtin_prefetch(kg + 32 * H_ * D_, 0, 0);
                __builtin_prefetch(vg + 32 * H_ * D_, 0, 0);
            }
        }
        __syncthreads();

        if (key_base > q0 + 15) continue;        // fully masked for this wave

        // wave-uniform: only diagonal blocks need per-element masking
        if (key_base + 31 > q0) block_body(key_base, true);
        else                    block_body(key_base, false);
    }

    // ---- epilogue: reduce per-lane l across the 16-lane half, O = acc/l --
    #pragma unroll
    for (int r = 0; r < 8; ++r) {
        const float lr  = half_reduce_sum(lrow[r]);
        const float inv = 1.0f / lr;
        const int row = q0 + r + 8 * half;
        float* orow = O + (((size_t)b * L_ + row) * H_ + h) * D_;
        orow[hn]      = acc[0][r] * inv;
        orow[16 + hn] = acc[1][r] * inv;
        orow[32 + hn] = acc[2][r] * inv;
        orow[48 + hn] = acc[3][r] * inv;
    }
}

extern "C" void kernel_launch(void* const* d_in, const int* in_sizes, int n_in,
                              void* d_out, int out_size, void* d_ws, size_t ws_size,
                              hipStream_t stream) {
    (void)in_sizes; (void)n_in; (void)d_ws; (void)ws_size; (void)out_size;
    const float* Q     = (const float*)d_in[0];
    const float* K     = (const float*)d_in[1];
    const float* V     = (const float*)d_in[2];
    // d_in[3] = attn_mask (bool, causal triu) -- applied analytically
    const float* tau   = (const float*)d_in[4];
    const float* delta = (const float*)d_in[5];
    float* O = (float*)d_out;

    dim3 grid(L_ / 64, H_, B_);   // (16, 16, 8)
    dim3 block(128);
    dsattn_fwd_kernel<<<grid, block, 0, stream>>>(Q, K, V, tau, delta, O);
}
